// MinGRUCell_11957188952157
// MI455X (gfx1250) — matrix operand order repaired and checked by reading, not compiled, not run on hardware
//
#include <hip/hip_runtime.h>
#include <hip/hip_bf16.h>
#include <stdint.h>

#define Bn 8
#define Tn 4096
#define Dn 256
#define Hn 256
#define MTOT (Bn * Tn)   // 32768 flattened rows

typedef __attribute__((ext_vector_type(16))) __bf16       v16bf;
typedef __attribute__((ext_vector_type(8)))  float        v8f;
typedef __attribute__((ext_vector_type(4)))  __bf16       v4bf;
typedef __attribute__((ext_vector_type(4)))  float        v4f;
typedef __attribute__((ext_vector_type(4)))  unsigned int v4u;

// Types matching the async-LDS builtin's parameters:
//   param0: int4 in AS1 (prints as "__device__" in HIP diagnostics)
//   param1: int4 in AS3 ("__shared__")
typedef int v4i_gcc __attribute__((vector_size(16)));
typedef __attribute__((address_space(1))) v4i_gcc as1_v4i;
typedef __attribute__((address_space(3))) v4i_gcc as3_v4i;

union Frag {
  v16bf v;
  struct { v4u lo; v4u hi; } q;
};

#if defined(__gfx1250__) && __has_builtin(__builtin_amdgcn_global_load_async_to_lds_b128)
#define HAVE_ASYNC_LDS 1
#else
#define HAVE_ASYNC_LDS 0
#endif

// ---------------------------------------------------------------------------
// Kernel 0: fp32 -> bf16 conversion (weights once, x once; x is the only
// streaming operand so bf16 halves the GEMM's HBM read).
// ---------------------------------------------------------------------------
__global__ __launch_bounds__(256) void mingru_cvt_bf16(
    const float* __restrict__ src, __bf16* __restrict__ dst, int n4) {
  int i = blockIdx.x * blockDim.x + threadIdx.x;
  if (i < n4) {
    v4f f = ((const v4f*)src)[i];
    v4bf o;
    o[0] = (__bf16)f[0];
    o[1] = (__bf16)f[1];
    o[2] = (__bf16)f[2];
    o[3] = (__bf16)f[3];
    ((v4bf*)dst)[i] = o;
  }
}

// ---------------------------------------------------------------------------
// Kernel 1: fused gate/update GEMM with bf16 WMMA, fp32 accumulate.
//   gate   = sigmoid(x @ Wz^T + bz)   -> G
//   update =          x @ Wh^T + bh   -> U
// Block = 256 thr = 8 waves; M-tile = 32 rows; x tile staged in LDS via
// async global->LDS (fallback: ds_store).
// Wave mapping: target = w&1 (gate/update), msub = (w>>1)&1 (16-row half),
//               nhalf = (w>>2)&1 (128-col half) -> 8 WMMA acc tiles / wave.
// ---------------------------------------------------------------------------
__global__ __launch_bounds__(256, 1) void mingru_gemm_wmma(
    const __bf16* __restrict__ xbf,
    const __bf16* __restrict__ wzb, const __bf16* __restrict__ whb,
    const float* __restrict__ bz, const float* __restrict__ bh,
    float* __restrict__ G, float* __restrict__ U) {
  __shared__ __bf16 lds_tile[32 * Dn];   // 16 KB x-tile, rows row0..row0+31

  const int tid = threadIdx.x;
  const size_t row0 = (size_t)blockIdx.x * 32;

  // ---- Stage x tile: contiguous 16 KB global -> LDS --------------------
#if HAVE_ASYNC_LDS
  {
    const char* gsrc = (const char*)(xbf + row0 * Dn);
#pragma unroll
    for (int i = 0; i < 4; ++i) {
      unsigned off = (unsigned)(tid + i * 256) * 16u;
      __builtin_amdgcn_global_load_async_to_lds_b128(
          (as1_v4i*)(gsrc + off),
          (as3_v4i*)((char*)lds_tile + off),
          0, 0);
    }
#if __has_builtin(__builtin_amdgcn_s_wait_asynccnt)
    __builtin_amdgcn_s_wait_asynccnt(0);
#else
    asm volatile("s_wait_asynccnt 0" ::: "memory");
#endif
  }
#else
  {
    const v4u* gsrc = (const v4u*)(xbf + row0 * Dn);
#pragma unroll
    for (int i = 0; i < 4; ++i) {
      int e = tid + i * 256;           // 16-byte chunk index (1024 chunks)
      ((v4u*)lds_tile)[e] = gsrc[e];
    }
  }
#endif
  __syncthreads();

  // ---- Wave/lane mapping ----------------------------------------------
  const int lane = tid & 31;
  const int wv = tid >> 5;
  const int target = wv & 1;         // 0 = gate(Wz,bz), 1 = update(Wh,bh)
  const int msub = (wv >> 1) & 1;    // which 16-row half of the 32-row tile
  const int nh = (wv >> 2) & 1;      // which 128-col half of H
  const int l15 = lane & 15;
  const int kh = lane >> 4;          // K-half selector per WMMA layout

  const __bf16* __restrict__ W = target ? whb : wzb;
  const float* __restrict__ bias = target ? bh : bz;
  float* __restrict__ OUT = target ? U : G;

  v8f zero = {0.f, 0.f, 0.f, 0.f, 0.f, 0.f, 0.f, 0.f};
  v8f acc[8];
#pragma unroll
  for (int nt = 0; nt < 8; ++nt) acc[nt] = zero;

  // A-row base in LDS (elements): lane holds row (msub*16 + l15)
  const int arow = (msub * 16 + l15) * Dn;

#pragma unroll
  for (int kk = 0; kk < 8; ++kk) {      // K loop: D=256, 32 per WMMA
    const int k0 = kk * 32;
    // A frag 16x32 bf16: lanes 0-15 K=[k0..k0+7]&[k0+16..k0+23],
    //                    lanes16-31 K=[k0+8..]&[k0+24..]
    Frag A;
    A.q.lo = *(const v4u*)(lds_tile + arow + k0 + kh * 8);
    A.q.hi = *(const v4u*)(lds_tile + arow + k0 + kh * 8 + 16);
#pragma unroll
    for (int nt = 0; nt < 8; ++nt) {
      const int h = nh * 128 + nt * 16 + l15;
      // B frag 32x16: lane holds column h, K = k0+kh*8..+7 & k0+16+kh*8..+7
      // B[k][n] = W[n][k] -> two 8-bf16 runs (16 apart) of row-major W[h][*]
      const __bf16* wp = W + (size_t)h * Dn + k0 + kh * 8;
      Frag Bm;
      Bm.q.lo = *(const v4u*)(wp);
      Bm.q.hi = *(const v4u*)(wp + 16);
      acc[nt] = __builtin_amdgcn_wmma_f32_16x16x32_bf16(
          false, A.v, false, Bm.v, (short)0, acc[nt], false, false);
    }
  }

  // ---- Epilogue: +bias, sigmoid for gate (wave-uniform branch) ---------
  const size_t rbase = row0 + (size_t)msub * 16 + (size_t)kh * 8;
#pragma unroll
  for (int nt = 0; nt < 8; ++nt) {
    const int h = nh * 128 + nt * 16 + l15;
    const float bv = bias[h];
#pragma unroll
    for (int r = 0; r < 8; ++r) {   // C/D layout: M = r + 8*kh, N = l15
      float v = acc[nt][r] + bv;
      if (target == 0) v = 1.0f / (1.0f + __expf(-v));
      OUT[(rbase + r) * Hn + h] = v;
    }
  }
}

// ---------------------------------------------------------------------------
// Kernel 2: sequential scan h = (1-g)h + g*u over T, one thread per (b,h).
// 2048 chains; loads are h-independent so unroll-4 batches them ahead of the
// dependent FMA chain. Fully coalesced (lane = h).
// ---------------------------------------------------------------------------
__global__ __launch_bounds__(256) void mingru_scan(
    const float* __restrict__ G, const float* __restrict__ U,
    const float* __restrict__ h0, float* __restrict__ out) {
  const int idx = blockIdx.x * 256 + threadIdx.x;   // 0 .. B*H-1
  const int b = idx >> 8;                           // H = 256
  float hv = h0[idx];

  const size_t base = (size_t)b * Tn * Hn + (idx & 255);
  const float* gp = G + base;
  const float* up = U + base;
  float* op = out + base;

  for (int t = 0; t < Tn; t += 4) {
    __builtin_prefetch(gp + 16 * Hn, 0, 0);
    __builtin_prefetch(up + 16 * Hn, 0, 0);
    float g0 = gp[0],      g1 = gp[Hn],     g2 = gp[2 * Hn], g3 = gp[3 * Hn];
    float u0 = up[0],      u1 = up[Hn],     u2 = up[2 * Hn], u3 = up[3 * Hn];
    hv = fmaf(g0, u0 - hv, hv); op[0] = hv;
    hv = fmaf(g1, u1 - hv, hv); op[Hn] = hv;
    hv = fmaf(g2, u2 - hv, hv); op[2 * Hn] = hv;
    hv = fmaf(g3, u3 - hv, hv); op[3 * Hn] = hv;
    gp += 4 * Hn; up += 4 * Hn; op += 4 * Hn;
  }
}

// ---------------------------------------------------------------------------
extern "C" void kernel_launch(void* const* d_in, const int* in_sizes, int n_in,
                              void* d_out, int out_size, void* d_ws, size_t ws_size,
                              hipStream_t stream) {
  const float* x  = (const float*)d_in[0];   // [B,T,D]
  const float* h0 = (const float*)d_in[1];   // [B,H]
  const float* Wz = (const float*)d_in[2];   // [H,D]
  const float* bz = (const float*)d_in[3];   // [H]
  const float* Wh = (const float*)d_in[4];   // [H,D]
  const float* bh = (const float*)d_in[5];   // [H]
  float* out = (float*)d_out;                // [B,T,H]

  // Workspace layout (all offsets 16B-aligned):
  char* ws = (char*)d_ws;
  __bf16* wzb = (__bf16*)(ws);                         // 131072 B
  __bf16* whb = (__bf16*)(ws + 131072);                // 131072 B
  __bf16* xb  = (__bf16*)(ws + 262144);                // 16 MiB
  float*  G   = (float*)(ws + 262144 + (size_t)MTOT * Dn * 2);              // 32 MiB
  float*  U   = (float*)(ws + 262144 + (size_t)MTOT * Dn * 2
                            + (size_t)MTOT * Hn * 4);                        // 32 MiB

  // 0) fp32 -> bf16 (weights + x)
  mingru_cvt_bf16<<<64, 256, 0, stream>>>(Wz, wzb, (Hn * Dn) / 4);
  mingru_cvt_bf16<<<64, 256, 0, stream>>>(Wh, whb, (Hn * Dn) / 4);
  mingru_cvt_bf16<<<(MTOT * Dn / 4 + 255) / 256, 256, 0, stream>>>(
      x, xb, (MTOT * Dn) / 4);

  // 1) fused gate/update GEMM (WMMA bf16)
  mingru_gemm_wmma<<<MTOT / 32, 256, 0, stream>>>(xb, wzb, whb, bz, bh, G, U);

  // 2) sequential scan
  mingru_scan<<<(Bn * Hn) / 256, 256, 0, stream>>>(G, U, h0, out);
}
